// TransformerLayer_88502096101922
// MI455X (gfx1250) — compile-verified, matrix operand order checked by reading, not run on hardware
//
#include <hip/hip_runtime.h>
#include <math.h>

// ---------------------------------------------------------------------------
// CDNA5 (gfx1250) wave32 WMMA types & helpers
// ---------------------------------------------------------------------------
typedef __attribute__((ext_vector_type(16))) __bf16 v16bf;
typedef __attribute__((ext_vector_type(8)))  __bf16 v8bf;
typedef __attribute__((ext_vector_type(8)))  float  v8f;

union Frag16 { uint4 u2[2]; v16bf v; };

__device__ __forceinline__ v8bf cvt8(float4 a, float4 b) {
    v8bf p;
    p[0] = (__bf16)a.x; p[1] = (__bf16)a.y; p[2] = (__bf16)a.z; p[3] = (__bf16)a.w;
    p[4] = (__bf16)b.x; p[5] = (__bf16)b.y; p[6] = (__bf16)b.z; p[7] = (__bf16)b.w;
    return p;
}

__device__ __forceinline__ v8f wmma_bf16(v16bf a, v16bf b, v8f c) {
    // D(f32 16x16) = A(bf16 16x32) * B(bf16 32x16) + C
    return __builtin_amdgcn_wmma_f32_16x16x32_bf16(false, a, false, b,
                                                   (short)0, c, false, false);
}

// ---------------------------------------------------------------------------
// Generic GEMM:  C[m,n] = alpha * ( concat(A0,A1)[m,:] . W[n,:] + bias[n] ) (+ resid[m,n])
// A0: first K0 columns, A1: remaining K-K0 columns (may be null).
// W row-major [N,K].  BM=128, BN=64, BK=32, 256 threads = 8 wave32.
// ---------------------------------------------------------------------------
#define GBM 128
#define GBN 64
#define GBK 32

__global__ __launch_bounds__(256)
void gemm_bias_kernel(const float* __restrict__ A0, const float* __restrict__ A1, int K0,
                      const float* __restrict__ W,  const float* __restrict__ bias,
                      const float* __restrict__ resid, float alpha,
                      float* __restrict__ C, int M, int N, int K)
{
    __shared__ __bf16 As[GBM][GBK];   // 8 KB
    __shared__ __bf16 Ws[GBN][GBK];   // 4 KB

    const int t    = threadIdx.x;
    const int lane = t & 31;
    const int wave = t >> 5;
    const int half = lane >> 4;
    const int l16  = lane & 15;
    const int kb   = half * 8;               // element offset of this half-wave's K chunk

    const int mBase = blockIdx.y * GBM;
    const int nBase = blockIdx.x * GBN;

    v8f acc[4];
    #pragma unroll
    for (int i = 0; i < 4; ++i)
        #pragma unroll
        for (int j = 0; j < 8; ++j) acc[i][j] = 0.0f;

    const int aRow = t >> 1;                 // 0..127
    const int aCol = (t & 1) << 4;           // 0 or 16
    const int wRow = t >> 2;                 // 0..63
    const int wCol = (t & 3) << 3;           // 0,8,16,24

    for (int kk0 = 0; kk0 < K; kk0 += GBK) {
        // ---- stage A tile (f32 -> bf16), handles the concat split ----
        const float* src; int ld, cb;
        if (kk0 < K0) { src = A0; ld = K0;     cb = kk0;      }
        else          { src = A1; ld = K - K0; cb = kk0 - K0; }
        const float* ap = src + (size_t)(mBase + aRow) * ld + cb + aCol;
        {
            float4 f0 = *(const float4*)(ap);
            float4 f1 = *(const float4*)(ap + 4);
            float4 f2 = *(const float4*)(ap + 8);
            float4 f3 = *(const float4*)(ap + 12);
            *(v8bf*)&As[aRow][aCol]     = cvt8(f0, f1);
            *(v8bf*)&As[aRow][aCol + 8] = cvt8(f2, f3);
        }
        if (kk0 + GBK < K) __builtin_prefetch(ap + GBK, 0, 0);

        // ---- stage W tile ----
        const float* wp = W + (size_t)(nBase + wRow) * K + kk0 + wCol;
        {
            float4 f0 = *(const float4*)(wp);
            float4 f1 = *(const float4*)(wp + 4);
            *(v8bf*)&Ws[wRow][wCol] = cvt8(f0, f1);
        }
        __syncthreads();

        // ---- WMMA: A frag 16x32 per wave, 4 column tiles of B ----
        const int ar = wave * 16 + l16;
        Frag16 af;
        af.u2[0] = *(const uint4*)&As[ar][kb];
        af.u2[1] = *(const uint4*)&As[ar][16 + kb];
        #pragma unroll
        for (int ct = 0; ct < 4; ++ct) {
            const int wr = ct * 16 + l16;
            Frag16 bf;
            bf.u2[0] = *(const uint4*)&Ws[wr][kb];
            bf.u2[1] = *(const uint4*)&Ws[wr][16 + kb];
            acc[ct] = wmma_bf16(af.v, bf.v, acc[ct]);
        }
        __syncthreads();
    }

    // ---- epilogue: alpha*(acc+bias) (+resid) ----
    #pragma unroll
    for (int ct = 0; ct < 4; ++ct) {
        const int n  = nBase + ct * 16 + l16;
        const float bv = bias[n];
        #pragma unroll
        for (int v = 0; v < 8; ++v) {
            const int m = mBase + wave * 16 + half * 8 + v;
            float val = alpha * (acc[ct][v] + bv);
            if (resid) val += resid[(size_t)m * N + n];
            C[(size_t)m * N + n] = val;
        }
    }
}

// ---------------------------------------------------------------------------
// Flash attention: per (b,h), Q[N,64] . K[N,64]^T -> online softmax -> . V[N,64]
// Layout of Q/K/V/O: [B, N, 256] with head h occupying columns h*64 .. h*64+63.
// Block = 128 query rows (8 wave32 x 16 rows), streams key blocks of 32.
// ---------------------------------------------------------------------------
#define FQ 128
#define FK 32

__global__ __launch_bounds__(256)
void flash_attn_kernel(const float* __restrict__ Q, const float* __restrict__ K,
                       const float* __restrict__ V, float* __restrict__ O,
                       float scale, int N)
{
    __shared__ __bf16 Qs[FQ][64];        // 16 KB
    __shared__ __bf16 Ks[FK][64];        //  4 KB
    __shared__ __bf16 Vts[64][FK];       //  4 KB (transposed)
    __shared__ __bf16 Ps[8][16][FK];     //  8 KB (per-wave P scratch)

    const int t    = threadIdx.x;
    const int lane = t & 31;
    const int wave = t >> 5;
    const int half = lane >> 4;
    const int l16  = lane & 15;
    const int kb   = half * 8;

    const int b = blockIdx.z, h = blockIdx.y;
    const int qBase = blockIdx.x * FQ;
    const size_t hoff = (size_t)b * N * 256 + (size_t)h * 64;

    // ---- stage Q tile (128x64) as bf16 ----
    {
        const int row = t >> 1;
        const int cb  = (t & 1) << 5;
        const float* qp = Q + hoff + (size_t)(qBase + row) * 256 + cb;
        #pragma unroll
        for (int j = 0; j < 4; ++j) {
            float4 f0 = *(const float4*)(qp + j * 8);
            float4 f1 = *(const float4*)(qp + j * 8 + 4);
            *(v8bf*)&Qs[row][cb + j * 8] = cvt8(f0, f1);
        }
    }
    __syncthreads();

    const int qr = wave * 16 + l16;
    Frag16 qa0, qa1;                              // Q fragments, hd k-steps 0..31 / 32..63
    qa0.u2[0] = *(const uint4*)&Qs[qr][kb];
    qa0.u2[1] = *(const uint4*)&Qs[qr][16 + kb];
    qa1.u2[0] = *(const uint4*)&Qs[qr][32 + kb];
    qa1.u2[1] = *(const uint4*)&Qs[qr][48 + kb];

    v8f o[4];
    float mrow[8], lrow[8];
    #pragma unroll
    for (int i = 0; i < 4; ++i)
        #pragma unroll
        for (int j = 0; j < 8; ++j) o[i][j] = 0.0f;
    #pragma unroll
    for (int v = 0; v < 8; ++v) { mrow[v] = -1.0e30f; lrow[v] = 0.0f; }

    for (int nk = 0; nk < N; nk += FK) {
        // ---- stage K block (32x64) and V block transposed (64x32) ----
        {
            const int row = t >> 3;               // 0..31
            const int cb  = (t & 7) << 3;         // 0..56
            const float* kp = K + hoff + (size_t)(nk + row) * 256 + cb;
            {
                float4 f0 = *(const float4*)(kp);
                float4 f1 = *(const float4*)(kp + 4);
                *(v8bf*)&Ks[row][cb] = cvt8(f0, f1);
            }
            const float* vp = V + hoff + (size_t)(nk + row) * 256 + cb;
            float4 v0f = *(const float4*)(vp);
            float4 v1f = *(const float4*)(vp + 4);
            Vts[cb + 0][row] = (__bf16)v0f.x; Vts[cb + 1][row] = (__bf16)v0f.y;
            Vts[cb + 2][row] = (__bf16)v0f.z; Vts[cb + 3][row] = (__bf16)v0f.w;
            Vts[cb + 4][row] = (__bf16)v1f.x; Vts[cb + 5][row] = (__bf16)v1f.y;
            Vts[cb + 6][row] = (__bf16)v1f.z; Vts[cb + 7][row] = (__bf16)v1f.w;
        }
        __syncthreads();

        // ---- S = scale * Q.K^T : two 16x16 fragments, 2 hd-ksteps each ----
        v8f s0, s1;
        #pragma unroll
        for (int j = 0; j < 8; ++j) { s0[j] = 0.0f; s1[j] = 0.0f; }
        {
            Frag16 bf;
            bf.u2[0] = *(const uint4*)&Ks[l16][kb];
            bf.u2[1] = *(const uint4*)&Ks[l16][16 + kb];
            s0 = wmma_bf16(qa0.v, bf.v, s0);
            bf.u2[0] = *(const uint4*)&Ks[l16][32 + kb];
            bf.u2[1] = *(const uint4*)&Ks[l16][48 + kb];
            s0 = wmma_bf16(qa1.v, bf.v, s0);
            bf.u2[0] = *(const uint4*)&Ks[16 + l16][kb];
            bf.u2[1] = *(const uint4*)&Ks[16 + l16][16 + kb];
            s1 = wmma_bf16(qa0.v, bf.v, s1);
            bf.u2[0] = *(const uint4*)&Ks[16 + l16][32 + kb];
            bf.u2[1] = *(const uint4*)&Ks[16 + l16][48 + kb];
            s1 = wmma_bf16(qa1.v, bf.v, s1);
        }
        #pragma unroll
        for (int v = 0; v < 8; ++v) { s0[v] *= scale; s1[v] *= scale; }

        // ---- online softmax (row reductions across the 16-lane half-wave) ----
        float corr[8];
        #pragma unroll
        for (int v = 0; v < 8; ++v) {
            float rm = fmaxf(s0[v], s1[v]);
            #pragma unroll
            for (int sh = 1; sh < 16; sh <<= 1)
                rm = fmaxf(rm, __shfl_xor(rm, sh, 32));
            const float mn = fmaxf(mrow[v], rm);
            corr[v] = __expf(mrow[v] - mn);
            mrow[v] = mn;
            const float p0 = __expf(s0[v] - mn);
            const float p1 = __expf(s1[v] - mn);
            s0[v] = p0; s1[v] = p1;
            float ps = p0 + p1;
            #pragma unroll
            for (int sh = 1; sh < 16; sh <<= 1)
                ps += __shfl_xor(ps, sh, 32);
            lrow[v] = lrow[v] * corr[v] + ps;
        }
        #pragma unroll
        for (int ct = 0; ct < 4; ++ct)
            #pragma unroll
            for (int v = 0; v < 8; ++v) o[ct][v] *= corr[v];

        // ---- D-layout P -> A-layout via per-wave LDS round-trip ----
        #pragma unroll
        for (int v = 0; v < 8; ++v) {
            const int pr = half * 8 + v;
            Ps[wave][pr][l16]      = (__bf16)s0[v];
            Ps[wave][pr][16 + l16] = (__bf16)s1[v];
        }
        Frag16 pa;
        pa.u2[0] = *(const uint4*)&Ps[wave][l16][kb];
        pa.u2[1] = *(const uint4*)&Ps[wave][l16][16 + kb];

        // ---- O += P.V  (4 column tiles of hd) ----
        #pragma unroll
        for (int ct = 0; ct < 4; ++ct) {
            Frag16 bf;
            const int vc = ct * 16 + l16;
            bf.u2[0] = *(const uint4*)&Vts[vc][kb];
            bf.u2[1] = *(const uint4*)&Vts[vc][16 + kb];
            o[ct] = wmma_bf16(pa.v, bf.v, o[ct]);
        }
        __syncthreads();
    }

    // ---- normalize and store (merged [B,N,D] layout) ----
    #pragma unroll
    for (int ct = 0; ct < 4; ++ct) {
        #pragma unroll
        for (int v = 0; v < 8; ++v) {
            const int m = qBase + wave * 16 + half * 8 + v;
            O[hoff + (size_t)m * 256 + ct * 16 + l16] = o[ct][v] / lrow[v];
        }
    }
}

// ---------------------------------------------------------------------------
// RoPE + QKV split:  qkv [M, 3*256] innermost-interleaved (h,d,j) -> roped Q,K and V
// ---------------------------------------------------------------------------
__global__ __launch_bounds__(256)
void rope_split_kernel(const float* __restrict__ qkv, const float* __restrict__ enc,
                       float* __restrict__ Qo, float* __restrict__ Ko,
                       float* __restrict__ Vo, int encStride)
{
    const int idx = blockIdx.x * 256 + threadIdx.x;   // over M*256
    const int m   = idx >> 8;
    const int col = idx & 255;
    const int d   = col & 63;

    const float f0 = enc[(size_t)m * 64 + d];
    const float f1 = enc[(size_t)encStride + (size_t)m * 64 + d];

    const size_t base  = (size_t)m * 768 + (size_t)col * 3;
    const int   pcol   = (d & 1) ? (col - 1) : (col + 1);
    const size_t pbase = (size_t)m * 768 + (size_t)pcol * 3;

    const float tq = qkv[base + 0], tk = qkv[base + 1], tv = qkv[base + 2];
    const float pq = qkv[pbase + 0], pk = qkv[pbase + 1];
    const float rq = (d & 1) ? pq : -pq;    // rotate_half
    const float rk = (d & 1) ? pk : -pk;

    Qo[idx] = tq * f0 + rq * f1;
    Ko[idx] = tk * f0 + rk * f1;
    Vo[idx] = tv;
}

// ---------------------------------------------------------------------------
// LayerNorm (512-wide) + exact GELU, one row per block
// ---------------------------------------------------------------------------
__global__ __launch_bounds__(256)
void ln_gelu_kernel(const float* __restrict__ X, const float* __restrict__ g,
                    const float* __restrict__ bb, float* __restrict__ Y)
{
    __shared__ float r1[256], r2[256];
    const int row = blockIdx.x;
    const int t   = threadIdx.x;
    const float* x = X + (size_t)row * 512;

    const float a0 = x[t], a1 = x[t + 256];
    r1[t] = a0 + a1;
    r2[t] = a0 * a0 + a1 * a1;
    __syncthreads();
    for (int off = 128; off > 0; off >>= 1) {
        if (t < off) { r1[t] += r1[t + off]; r2[t] += r2[t + off]; }
        __syncthreads();
    }
    const float mean = r1[0] * (1.0f / 512.0f);
    const float var  = r2[0] * (1.0f / 512.0f) - mean * mean;
    const float inv  = rsqrtf(var + 1e-5f);

    const float y0 = (a0 - mean) * inv * g[t]       + bb[t];
    const float y1 = (a1 - mean) * inv * g[t + 256] + bb[t + 256];
    Y[(size_t)row * 512 + t]       = 0.5f * y0 * (1.0f + erff(y0 * 0.70710678118654752f));
    Y[(size_t)row * 512 + t + 256] = 0.5f * y1 * (1.0f + erff(y1 * 0.70710678118654752f));
}

// ---------------------------------------------------------------------------
// Orchestration
// ---------------------------------------------------------------------------
extern "C" void kernel_launch(void* const* d_in, const int* in_sizes, int n_in,
                              void* d_out, int out_size, void* d_ws, size_t ws_size,
                              hipStream_t stream)
{
    (void)in_sizes; (void)n_in; (void)out_size; (void)ws_size;
    const int B = 8, Nn = 2048, D = 256, M = B * Nn;   // M = 16384

    const float* desc0  = (const float*)d_in[0];
    const float* desc1  = (const float*)d_in[1];
    const float* enc0   = (const float*)d_in[2];
    const float* enc1   = (const float*)d_in[3];
    const float* Wqkv_w = (const float*)d_in[4];
    const float* Wqkv_b = (const float*)d_in[5];
    const float* out_w  = (const float*)d_in[6];
    const float* out_b  = (const float*)d_in[7];
    const float* sf1_w  = (const float*)d_in[8];
    const float* sf1_b  = (const float*)d_in[9];
    const float* sln_g  = (const float*)d_in[10];
    const float* sln_b  = (const float*)d_in[11];
    const float* sf2_w  = (const float*)d_in[12];
    const float* sf2_b  = (const float*)d_in[13];
    const float* qk_w   = (const float*)d_in[14];
    const float* qk_b   = (const float*)d_in[15];
    const float* v_w    = (const float*)d_in[16];
    const float* v_b    = (const float*)d_in[17];
    const float* co_w   = (const float*)d_in[18];
    const float* co_b   = (const float*)d_in[19];
    const float* cf1_w  = (const float*)d_in[20];
    const float* cf1_b  = (const float*)d_in[21];
    const float* cln_g  = (const float*)d_in[22];
    const float* cln_b  = (const float*)d_in[23];
    const float* cf2_w  = (const float*)d_in[24];
    const float* cf2_b  = (const float*)d_in[25];

    float* ws  = (float*)d_ws;
    float* qkv = ws;                               // M*768
    float* qb  = qkv + (size_t)M * 768;            // M*256
    float* kb  = qb  + (size_t)M * 256;            // M*256
    float* vb  = kb  + (size_t)M * 256;            // M*256
    float* ctx = vb  + (size_t)M * 256;            // M*256
    float* msg = ctx + (size_t)M * 256;            // M*256
    float* hb  = msg + (size_t)M * 256;            // M*512
    float* gb  = hb  + (size_t)M * 512;            // M*512
    float* d0  = gb  + (size_t)M * 512;            // M*256
    float* d1  = d0  + (size_t)M * 256;            // M*256

    const dim3 blk(256);
    auto gemm = [&](const float* A0, const float* A1, int K0, const float* W,
                    const float* bias, const float* resid, float alpha,
                    float* C, int N, int K) {
        dim3 grid(N / GBN, M / GBM);
        gemm_bias_kernel<<<grid, blk, 0, stream>>>(A0, A1, K0, W, bias, resid, alpha, C, M, N, K);
    };
    auto flash = [&](const float* Q, const float* K, const float* V, float* O, float scale) {
        flash_attn_kernel<<<dim3(Nn / FQ, 4, B), blk, 0, stream>>>(Q, K, V, O, scale, Nn);
    };

    auto self_block = [&](const float* x, const float* enc, float* dout) {
        gemm(x, nullptr, D, Wqkv_w, Wqkv_b, nullptr, 1.0f, qkv, 768, 256);
        rope_split_kernel<<<dim3(M), blk, 0, stream>>>(qkv, enc, qb, kb, vb, M * 64);
        flash(qb, kb, vb, ctx, 0.125f);                          // hd^-0.5
        gemm(ctx, nullptr, D, out_w, out_b, nullptr, 1.0f, msg, 256, 256);
        gemm(x, msg, D, sf1_w, sf1_b, nullptr, 1.0f, hb, 512, 512);
        ln_gelu_kernel<<<dim3(M), blk, 0, stream>>>(hb, sln_g, sln_b, gb);
        gemm(gb, nullptr, 512, sf2_w, sf2_b, x, 1.0f, dout, 256, 512);
    };

    self_block(desc0, enc0, d0);
    self_block(desc1, enc1, d1);

    // ---- cross block ----
    const float sqs = 0.35355339059327379f;        // (hd^-0.5)^0.5
    gemm(d0, nullptr, D, qk_w, qk_b, nullptr, sqs, qb, 256, 256);   // qk0
    gemm(d1, nullptr, D, qk_w, qk_b, nullptr, sqs, kb, 256, 256);   // qk1
    gemm(d0, nullptr, D, v_w,  v_b,  nullptr, 1.0f, vb, 256, 256);  // v0
    gemm(d1, nullptr, D, v_w,  v_b,  nullptr, 1.0f, msg, 256, 256); // v1

    float* ctx1 = qkv;                       // reuse qkv region: ctx1 | m0 | m1
    float* m0   = qkv + (size_t)M * 256;
    float* m1   = qkv + (size_t)M * 512;
    flash(qb, kb, msg, ctx, 1.0f);           // m0 raw = softmax_j(qk0.qk1^T).v1
    flash(kb, qb, vb, ctx1, 1.0f);           // m1 raw = softmax_j(qk1.qk0^T).v0
    gemm(ctx,  nullptr, D, co_w, co_b, nullptr, 1.0f, m0, 256, 256);
    gemm(ctx1, nullptr, D, co_w, co_b, nullptr, 1.0f, m1, 256, 256);

    float* out0 = (float*)d_out;
    float* out1 = out0 + (size_t)M * 256;

    gemm(d0, m0, D, cf1_w, cf1_b, nullptr, 1.0f, hb, 512, 512);
    ln_gelu_kernel<<<dim3(M), blk, 0, stream>>>(hb, cln_g, cln_b, gb);
    gemm(gb, nullptr, 512, cf2_w, cf2_b, d0, 1.0f, out0, 256, 512);

    gemm(d1, m1, D, cf1_w, cf1_b, nullptr, 1.0f, hb, 512, 512);
    ln_gelu_kernel<<<dim3(M), blk, 0, stream>>>(hb, cln_g, cln_b, gb);
    gemm(gb, nullptr, 512, cf2_w, cf2_b, d1, 1.0f, out1, 256, 512);
}
